// DogYOLOLoss_35708358099195
// MI455X (gfx1250) — compile-verified, matrix operand order checked by reading, not run on hardware
//
#include <hip/hip_runtime.h>
#include <hip/hip_bf16.h>
#include <math.h>

// ---------------------------------------------------------------------------
// DogYOLOLoss for MI455X (gfx1250, wave32).
// Memory/latency-bound reduction: only the obj channel (strided 596B gather,
// ~17MB of cachelines) + <=1152 winner cells matter => ~1us at 23.3 TB/s.
//  * v_wmma_f32_16x16x32_f16 is the wave32 reduction engine (hi/lo f16 split
//    of each partial, ones-B, f32 accumulate -> near-f32 accurate row sums).
//  * global_load_async_to_lds_b32 (ASYNCcnt) double-buffers the strided obj
//    gather through LDS so the wave never stalls on the gather latency.
//  * global_prefetch_b8 pulls each winner cell's first cacheline early.
// ---------------------------------------------------------------------------

#define IMG_DIM 640.0f
#define NB_CLS 120
#define NE_CLS 8
#define NA_CLS 12
#define BATCH  16
#define NBOX   24
#define CCH    (9 + NB_CLS + NE_CLS + NA_CLS)   // 149 channels
#define IGN_T  (-100)

typedef __attribute__((ext_vector_type(16))) _Float16 v16h;
typedef __attribute__((ext_vector_type(8)))  float    v8f;

// ---- wave32 sum via WMMA ---------------------------------------------------
// f16 A-matrix 16x32 layout (ISA 7.12.2): lane L<16 owns K=0,1 of row L in its
// v16h[0..1]; lane L>=16 owns K=8,9 of row L-16. Put the hi/lo f16 split of the
// partial there, B = ones => D[m][n] = sum_k A[m][k] with f32 accumulation, so
// row m already holds partial[m] + partial[m+16]. Lanes 0-15 hold rows 0-7 in
// c[0..7], lanes 16-31 hold rows 8-15; one shfl_xor(16) completes the sum.
__device__ __forceinline__ float wave_sum32(float x) {
  v16h a = {};
  _Float16 hi = (_Float16)x;
  a[0] = hi;
  a[1] = (_Float16)(x - (float)hi);   // residual: near-f32 accuracy overall
  v16h b;
  #pragma unroll
  for (int i = 0; i < 16; ++i) b[i] = (_Float16)1.0f;
  v8f c = {};
  c = __builtin_amdgcn_wmma_f32_16x16x32_f16(false, a, false, b,
                                             (short)0, c, false, false);
  float s = 0.f;
  #pragma unroll
  for (int i = 0; i < 8; ++i) s += c[i];
  s += __shfl_xor(s, 16, 32);         // combine rows 0-7 with rows 8-15
  return s;                           // every lane holds the full 32-lane sum
}

__device__ __forceinline__ float softplus_f(float x) {
  return fmaxf(x, 0.f) + log1pf(expf(-fabsf(x)));
}
__device__ __forceinline__ float sigmoid_f(float x) {
  return 1.f / (1.f + expf(-x));
}
__device__ __forceinline__ float smooth_l1(float p, float t) {
  float d = fabsf(p - t);
  return (d < 1.f) ? 0.5f * d * d : d - 0.5f;
}
__device__ __forceinline__ float safe_ce(const float* logits, int num, int tgt) {
  if (tgt == IGN_T) return 0.f;
  int t = min(max(tgt, 0), num - 1);
  float m = -INFINITY;
  for (int k = 0; k < num; ++k) m = fmaxf(m, logits[k]);
  float sum = 0.f;
  for (int k = 0; k < num; ++k) sum += expf(logits[k] - m);
  return m + logf(sum) - logits[t];
}

__device__ __forceinline__ unsigned lds_byte_off(void* p) {
  return (unsigned)(unsigned long long)(__attribute__((address_space(3))) void*)p;
}

// ---- kernel 1: sum softplus(obj) over all cells of all 3 scales ------------
__device__ __forceinline__ const float* obj_addr(const float* p0, const float* p1,
                                                 const float* p2, int i) {
  const int c0  = BATCH * 80 * 80;
  const int c01 = c0 + BATCH * 40 * 40;
  if (i < c0)  return p0 + (size_t)i * CCH;
  if (i < c01) return p1 + (size_t)(i - c0) * CCH;
  return p2 + (size_t)(i - c01) * CCH;
}

__global__ void k_obj_softplus(const float* __restrict__ p0,
                               const float* __restrict__ p1,
                               const float* __restrict__ p2,
                               float* __restrict__ blockSums) {
  __shared__ float abuf[2][256];     // async double buffer (one slot per thread)
  __shared__ float wsum[8];
  const int total  = BATCH * (80 * 80 + 40 * 40 + 20 * 20);  // 134400
  const int stride = gridDim.x * blockDim.x;
  const int tid    = threadIdx.x;
  float acc = 0.f;

  int i = blockIdx.x * blockDim.x + tid;
  int cur = 0;
  {   // prologue: stage first element into slot 0 (clamped; OOB lanes ignored)
    int ii = min(i, total - 1);
    unsigned lo = lds_byte_off(&abuf[cur][tid]);
    unsigned long long ga = (unsigned long long)obj_addr(p0, p1, p2, ii);
    asm volatile("global_load_async_to_lds_b32 %0, %1, off"
                 :: "v"(lo), "v"(ga) : "memory");
  }
  for (; i < total; i += stride) {
    {  // stage next chunk into the other slot while consuming this one
      int nx = min(i + stride, total - 1);
      unsigned lo = lds_byte_off(&abuf[cur ^ 1][tid]);
      unsigned long long ga = (unsigned long long)obj_addr(p0, p1, p2, nx);
      asm volatile("global_load_async_to_lds_b32 %0, %1, off"
                   :: "v"(lo), "v"(ga) : "memory");
    }
    asm volatile("s_wait_asynccnt 0x1" ::: "memory");  // current slot landed
    acc += softplus_f(abuf[cur][tid]);
    cur ^= 1;
  }
  asm volatile("s_wait_asynccnt 0x0" ::: "memory");    // drain before LDS reuse

  float w = wave_sum32(acc);
  int lane = tid & 31, wave = tid >> 5;
  if (lane == 0) wsum[wave] = w;
  __syncthreads();
  if (tid == 0) {
    float s = 0.f;
    #pragma unroll
    for (int k = 0; k < 8; ++k) s += wsum[k];
    blockSums[blockIdx.x] = s;   // L_OBJ = 1
  }
}

// ---- kernel 2: per-(scale,batch) box assignment + winner losses ------------
__global__ void k_per_box(const float* __restrict__ p0,
                          const float* __restrict__ p1,
                          const float* __restrict__ p2,
                          const float* __restrict__ body,
                          const float* __restrict__ headb,
                          const int*   __restrict__ labels,
                          const int*   __restrict__ emotions,
                          const int*   __restrict__ actions,
                          const unsigned char* __restrict__ head_valid,
                          float* __restrict__ lossPart,
                          float* __restrict__ posPart) {
  const int s = blockIdx.x / BATCH;
  const int b = blockIdx.x % BATCH;
  const int W = (s == 0) ? 80 : ((s == 1) ? 40 : 20);
  const int H = W;
  const float sx = IMG_DIM / (float)W, sy = IMG_DIM / (float)H;
  const float* pred = (s == 0) ? p0 : ((s == 1) ? p1 : p2);

  const int n = threadIdx.x;            // one wave32; lane = box index
  const bool active = n < NBOX;

  float bx1 = 0.f, by1 = 0.f, bx2 = 0.f, by2 = 0.f, area = -1.f;
  if (active) {
    const float* bp = body + ((size_t)b * NBOX + n) * 4;
    bx1 = bp[0]; by1 = bp[1]; bx2 = bp[2]; by2 = bp[3];
    // area uses the UNscaled (normalized) coords, per reference
    area = fmaxf(bx2 - bx1, 0.f) * fmaxf(by2 - by1, 0.f);
  }

  // rank = argsort(argsort(area)): #{j : area_j < area_i || (== && j < i)}
  int rank = 0;
  #pragma unroll
  for (int j = 0; j < NBOX; ++j) {
    float aj = __shfl(area, j, 32);
    rank += (aj < area) || (aj == area && j < n);
  }

  // scale boxes to pixels if normalized
  float mx = fmaxf(fmaxf(bx1, by1), fmaxf(bx2, by2));
  if (mx <= 1.5f) { bx1 *= IMG_DIM; by1 *= IMG_DIM; bx2 *= IMG_DIM; by2 *= IMG_DIM; }
  float bw = bx2 - bx1, bh = by2 - by1;
  float cx = 0.5f * (bx1 + bx2), cy = 0.5f * (by1 + by2);
  float gxf = floorf(cx / sx), gyf = floorf(cy / sy);
  int gx = (int)gxf, gy = (int)gyf;
  bool valid = active && (bw > 0.f) && (bh > 0.f) &&
               (gx >= 0) && (gy >= 0) && (gx < W) && (gy < H);
  int gxc = min(max(gx, 0), W - 1);
  int gyc = min(max(gy, 0), H - 1);
  int cellc = gyc * W + gxc;
  int reff = valid ? rank : (1000 + n);   // never beats a valid rank (< 24)

  const float* cp = pred + (((size_t)b * H + gyc) * (size_t)W + gxc) * CCH;
  if (valid) __builtin_prefetch(cp, 0, 1);   // pull cell line during ranking

  // winner: valid and no other box in the same cell with a smaller rank
  bool winner = valid;
  #pragma unroll
  for (int j = 0; j < 32; ++j) {
    int cj = __shfl(cellc, j, 32);
    int rj = __shfl(reff, j, 32);
    if (cj == cellc && rj < reff) winner = false;
  }

  float contrib = 0.f, posc = 0.f;
  if (winner) {
    posc = 1.f;
    // obj BCE winner-cell term: -obj * obj_target (L_OBJ = 1)
    contrib = -cp[0];
    // box loss (L_BOX = 5)
    float tx = cx / sx - gxf, ty = cy / sy - gyf;
    float boxl = smooth_l1(sigmoid_f(cp[1]), tx) +
                 smooth_l1(sigmoid_f(cp[2]), ty) +
                 smooth_l1(cp[3], logf(bw / sx + 1e-6f)) +
                 smooth_l1(cp[4], logf(bh / sy + 1e-6f));
    contrib += 5.0f * boxl;
    // head loss (L_HEAD = 2)
    const float* hp = headb + ((size_t)b * NBOX + n) * 4;
    float hx1 = hp[0], hy1 = hp[1], hx2 = hp[2], hy2 = hp[3];
    float hmx = fmaxf(fmaxf(hx1, hy1), fmaxf(hx2, hy2));
    if (hmx <= 1.5f) { hx1 *= IMG_DIM; hy1 *= IMG_DIM; hx2 *= IMG_DIM; hy2 *= IMG_DIM; }
    bool hv = head_valid[b * NBOX + n] != 0;
    if (hv && (hx2 > hx1) && (hy2 > hy1)) {
      float r0 = fminf(fmaxf((hx1 - bx1) / bw, 0.f), 1.f);
      float r1 = fminf(fmaxf((hy1 - by1) / bh, 0.f), 1.f);
      float r2 = fminf(fmaxf((hx2 - bx1) / bw, 0.f), 1.f);
      float r3 = fminf(fmaxf((hy2 - by1) / bh, 0.f), 1.f);
      float hl = smooth_l1(sigmoid_f(cp[5]), r0) + smooth_l1(sigmoid_f(cp[6]), r1) +
                 smooth_l1(sigmoid_f(cp[7]), r2) + smooth_l1(sigmoid_f(cp[8]), r3);
      contrib += 2.0f * hl;
    }
    // attribute CE (L_ATTR = 1)
    contrib += safe_ce(cp + 9,                   NB_CLS, labels  [b * NBOX + n]) +
               safe_ce(cp + 9 + NB_CLS,          NE_CLS, emotions[b * NBOX + n]) +
               safe_ce(cp + 9 + NB_CLS + NE_CLS, NA_CLS, actions [b * NBOX + n]);
  }

  // re-converged: EXEC is all-ones for the WMMA reductions
  float lsum = wave_sum32(contrib);
  float psum = wave_sum32(posc);
  if (threadIdx.x == 0) {
    lossPart[blockIdx.x] = lsum;
    posPart[blockIdx.x]  = psum;
  }
}

// ---- kernel 3: deterministic fixed-order final combine ---------------------
__global__ void k_finalize(const float* __restrict__ objPart, int nObj,
                           const float* __restrict__ lossPart,
                           const float* __restrict__ posPart, int nBox,
                           float* __restrict__ out) {
  if (blockIdx.x == 0 && threadIdx.x == 0) {
    float total = 0.f, pos = 0.f;
    for (int i = 0; i < nObj; ++i) total += objPart[i];
    for (int i = 0; i < nBox; ++i) { total += lossPart[i]; pos += posPart[i]; }
    out[0] = total / fmaxf(pos, 1.0f);
  }
}

extern "C" void kernel_launch(void* const* d_in, const int* in_sizes, int n_in,
                              void* d_out, int out_size, void* d_ws, size_t ws_size,
                              hipStream_t stream) {
  (void)in_sizes; (void)n_in; (void)out_size; (void)ws_size;
  const float* pred0  = (const float*)d_in[0];
  const float* pred1  = (const float*)d_in[1];
  const float* pred2  = (const float*)d_in[2];
  const float* body   = (const float*)d_in[3];
  const float* headb  = (const float*)d_in[4];
  const int*   labels = (const int*)d_in[5];
  const int*   emots  = (const int*)d_in[6];
  const int*   acts   = (const int*)d_in[7];
  const unsigned char* hvalid = (const unsigned char*)d_in[8];
  float* out = (float*)d_out;

  const int G_OBJ = 128;                     // ~4 pipelined iters/thread
  const int G_BOX = 3 * BATCH;               // 48 blocks, 1 wave32 each
  float* wsf      = (float*)d_ws;
  float* objPart  = wsf;                     // [128]
  float* lossPart = wsf + G_OBJ;             // [48]
  float* posPart  = wsf + G_OBJ + G_BOX;     // [48]  (fully overwritten each call)

  k_obj_softplus<<<G_OBJ, 256, 0, stream>>>(pred0, pred1, pred2, objPart);
  k_per_box<<<G_BOX, 32, 0, stream>>>(pred0, pred1, pred2, body, headb,
                                      labels, emots, acts, hvalid,
                                      lossPart, posPart);
  k_finalize<<<1, 32, 0, stream>>>(objPart, G_OBJ, lossPart, posPart, G_BOX, out);
}
